// HyperGraphSparseAttention_69037304316035
// MI455X (gfx1250) — matrix operand order repaired and checked by reading, not compiled, not run on hardware
//
#include <hip/hip_runtime.h>
#include <hip/hip_bf16.h>

#define DEVINL __device__ __forceinline__

typedef __bf16 bf16;
typedef __attribute__((ext_vector_type(16))) __bf16 v16bf;
typedef __attribute__((ext_vector_type(8)))  __bf16 v8bf;
typedef __attribute__((ext_vector_type(8)))  float  v8f;

constexpr int   Bsz = 2, Ssz = 2048, Esz = 1024, Hn = 16, Dh = 64, Kn = 4;
constexpr float SCL = 0.125f;                    // 64^-0.5
constexpr float LN_ROPE = 9.210340371976184f;    // ln(10000)

// ---------------- WMMA fragment helpers (ISA 7.12.2 layouts) ----------------

// A-matrix 16x32 bf16: lane m=L&15; half h=L>>4; elems 0..7 = K h*8..h*8+7,
// elems 8..15 = K 16+h*8 .. 16+h*8+7. Two contiguous 16B loads.
DEVINL v16bf load_a_frag(const bf16* base, int ld) {
  int lane = threadIdx.x & 31;
  int m = lane & 15, hf = lane >> 4;
  const bf16* p = base + (size_t)m * ld + hf * 8;
  v8bf lo = *(const v8bf*)(p);
  v8bf hi = *(const v8bf*)(p + 16);
  v16bf r;
#pragma unroll
  for (int i = 0; i < 8; ++i) { r[i] = lo[i]; r[8 + i] = hi[i]; }
  return r;
}

// B-matrix 32x16 bf16 from Bt[N][K] row-major: lane n=L&15; elems 0..7 = K h*8..h*8+7, elems 8..15 = K 16+h*8..16+h*8+7
DEVINL v16bf load_b_frag(const bf16* base, int ld) {
  int lane = threadIdx.x & 31;
  int n = lane & 15, hf = lane >> 4;
  const bf16* p = base + (size_t)n * ld + hf * 8;
  v8bf lo = *(const v8bf*)(p);
  v8bf hi = *(const v8bf*)(p + 16);
  v16bf r;
#pragma unroll
  for (int i = 0; i < 8; ++i) { r[i] = lo[i]; r[8 + i] = hi[i]; }
  return r;
}

DEVINL v8f wmma_bf16(v16bf a, v16bf b, v8f c) {
  return __builtin_amdgcn_wmma_f32_16x16x32_bf16(false, a, false, b, (short)0, c,
                                                 false, false);
}

DEVINL float rmax16(float v) {
#pragma unroll
  for (int off = 1; off < 16; off <<= 1) v = fmaxf(v, __shfl_xor(v, off, 16));
  return v;
}
DEVINL float rsum16(float v) {
#pragma unroll
  for (int off = 1; off < 16; off <<= 1) v += __shfl_xor(v, off, 16);
  return v;
}

// ---------------- elementwise conversions ----------------

__global__ __launch_bounds__(256) void k_f32_to_bf16(const float* __restrict__ in,
                                                     bf16* __restrict__ out, int n) {
  int i = blockIdx.x * 256 + threadIdx.x;
  if (i < n) out[i] = (bf16)in[i];
}

// dst[c][r] = (bf16)src[r][c]   (weights -> Bt layout)
__global__ __launch_bounds__(256) void k_transpose_cvt(const float* __restrict__ in,
                                                       bf16* __restrict__ out,
                                                       int rows, int cols) {
  int i = blockIdx.x * 256 + threadIdx.x;
  if (i < rows * cols) {
    int r = i / cols, c = i - r * cols;
    out[(size_t)c * rows + r] = (bf16)in[i];
  }
}

// ---------------- generic bf16 WMMA GEMM: C = A[MxK] * Bt[NxK]^T ----------------
// mode 0: scatter to [B,H,S,Dd] (m = b*S+s, n = h*Dd+d);  mode 1: row-major [M,N]
DEVINL void store_tile(float* __restrict__ C, v8f c, int m0, int n0, int N,
                       int mode, int Dd, int half, int col) {
#pragma unroll
  for (int r = 0; r < 8; ++r) {
    int m = m0 + half * 8 + r;
    int n = n0 + col;
    float vv = c[r];
    if (mode == 0) {
      int bb = m / Ssz, ss = m - bb * Ssz;
      int hh = n / Dd, dd = n - hh * Dd;
      C[(((size_t)bb * Hn + hh) * Ssz + ss) * (size_t)Dd + dd] = vv;
    } else {
      C[(size_t)m * N + n] = vv;
    }
  }
}

__global__ __launch_bounds__(256) void k_gemm_bf16(
    const bf16* __restrict__ A, const bf16* __restrict__ Bt, float* __restrict__ C,
    int M, int N, int Kd, int mode, int Dd) {
  int w = blockIdx.x * 8 + (threadIdx.x >> 5);
  int nt = N >> 5;
  int tm = w / nt, tn = w - tm * nt;
  int m0 = tm << 5, n0 = tn << 5;
  if (m0 >= M) return;

  v8f c00 = {}, c01 = {}, c10 = {}, c11 = {};
  for (int k = 0; k < Kd; k += 32) {
    v16bf a0 = load_a_frag(A + (size_t)m0 * Kd + k, Kd);
    v16bf a1 = load_a_frag(A + (size_t)(m0 + 16) * Kd + k, Kd);
    v16bf b0 = load_b_frag(Bt + (size_t)n0 * Kd + k, Kd);
    v16bf b1 = load_b_frag(Bt + (size_t)(n0 + 16) * Kd + k, Kd);
    c00 = wmma_bf16(a0, b0, c00);
    c01 = wmma_bf16(a0, b1, c01);
    c10 = wmma_bf16(a1, b0, c10);
    c11 = wmma_bf16(a1, b1, c11);
  }
  int lane = threadIdx.x & 31, half = lane >> 4, col = lane & 15;
  store_tile(C, c00, m0, n0, N, mode, Dd, half, col);
  store_tile(C, c01, m0, n0 + 16, N, mode, Dd, half, col);
  store_tile(C, c10, m0 + 16, n0, N, mode, Dd, half, col);
  store_tile(C, c11, m0 + 16, n0 + 16, N, mode, Dd, half, col);
}

// ---------------- routing softmax + cumsum + fractional RoPE ----------------
// one block per (b,h); sequential scan over S in chunks of 256 with carry.
__global__ __launch_bounds__(256) void k_route_rope(
    const float* __restrict__ Qf, const float* __restrict__ Kf,
    const float* __restrict__ Vf, const float* __restrict__ Lg,
    bf16* __restrict__ qrb, bf16* __restrict__ krb, bf16* __restrict__ vtb,
    float* __restrict__ Pb) {
  __shared__ float sc[4][256];
  int bh = blockIdx.x;
  int tid = threadIdx.x;
  float carry0 = 0.f, carry1 = 0.f, carry2 = 0.f, carry3 = 0.f;

  for (int s0 = 0; s0 < Ssz; s0 += 256) {
    int s = s0 + tid;
    const float* lg = Lg + ((size_t)bh * Ssz + s) * 4;
    float l0 = lg[0], l1 = lg[1], l2 = lg[2], l3 = lg[3];
    float mx = fmaxf(fmaxf(l0, l1), fmaxf(l2, l3));
    float e0 = __expf(l0 - mx), e1 = __expf(l1 - mx);
    float e2 = __expf(l2 - mx), e3 = __expf(l3 - mx);
    float inv = 1.f / (e0 + e1 + e2 + e3);
    float p0 = e0 * inv, p1 = e1 * inv, p2 = e2 * inv, p3 = e3 * inv;
    float* pp = Pb + ((size_t)bh * Ssz + s) * 4;
    pp[0] = p0; pp[1] = p1; pp[2] = p2; pp[3] = p3;

    sc[0][tid] = p0; sc[1][tid] = p1; sc[2][tid] = p2; sc[3][tid] = p3;
    __syncthreads();
    for (int off = 1; off < 256; off <<= 1) {
      float t0 = sc[0][tid], t1 = sc[1][tid], t2 = sc[2][tid], t3 = sc[3][tid];
      float u0 = 0.f, u1 = 0.f, u2 = 0.f, u3 = 0.f;
      if (tid >= off) {
        u0 = sc[0][tid - off]; u1 = sc[1][tid - off];
        u2 = sc[2][tid - off]; u3 = sc[3][tid - off];
      }
      __syncthreads();
      sc[0][tid] = t0 + u0; sc[1][tid] = t1 + u1;
      sc[2][tid] = t2 + u2; sc[3][tid] = t3 + u3;
      __syncthreads();
    }
    float pos0 = carry0 + sc[0][tid] - p0;
    float pos1 = carry1 + sc[1][tid] - p1;
    float pos2 = carry2 + sc[2][tid] - p2;
    float pos3 = carry3 + sc[3][tid] - p3;
    carry0 += sc[0][255]; carry1 += sc[1][255];
    carry2 += sc[2][255]; carry3 += sc[3][255];
    __syncthreads();

    pos0 = fminf(fmaxf(pos0, 0.f), 8191.f);
    pos1 = fminf(fmaxf(pos1, 0.f), 8191.f);
    pos2 = fminf(fmaxf(pos2, 0.f), 8191.f);
    pos3 = fminf(fmaxf(pos3, 0.f), 8191.f);

    const float* qp = Qf + ((size_t)bh * Ssz + s) * 64;
    const float* kp = Kf + ((size_t)bh * Ssz + s) * 64;
    const float* vp = Vf + ((size_t)bh * Ssz + s) * 64;
    bf16* qo = qrb + ((size_t)bh * Ssz + s) * 64;
    bf16* ko = krb + ((size_t)bh * Ssz + s) * 64;

    for (int d = 0; d < 32; ++d) {
      float invf = __expf((float)d * (-2.0f / 64.0f) * LN_ROPE);
      float cw = 0.f, sw = 0.f, sn, cs;
      __sincosf(pos0 * invf, &sn, &cs); cw += p0 * cs; sw += p0 * sn;
      __sincosf(pos1 * invf, &sn, &cs); cw += p1 * cs; sw += p1 * sn;
      __sincosf(pos2 * invf, &sn, &cs); cw += p2 * cs; sw += p2 * sn;
      __sincosf(pos3 * invf, &sn, &cs); cw += p3 * cs; sw += p3 * sn;
      float qa = qp[d], qb = qp[d + 32];
      qo[d]      = (bf16)(qa * cw - qb * sw);
      qo[d + 32] = (bf16)(qb * cw + qa * sw);
      float ka = kp[d], kb = kp[d + 32];
      ko[d]      = (bf16)(ka * cw - kb * sw);
      ko[d + 32] = (bf16)(kb * cw + ka * sw);
    }
    // V transposed per (b,h): vt[d][s], so PV B-fragments are contiguous in s.
    for (int d = 0; d < 64; ++d)
      vtb[((size_t)bh * 64 + d) * Ssz + s] = (bf16)vp[d];
  }
}

// ---------------- causal flash attention, one 16-query tile per wave ----------------
__global__ __launch_bounds__(256) void k_attn(
    const bf16* __restrict__ qr, const bf16* __restrict__ kr,
    const bf16* __restrict__ vt, const float* __restrict__ Pp,
    float* __restrict__ aout) {
  __shared__ bf16 ldsP[8 * 16 * 32];  // per-wave 16x32 exp(P) staging
  int wid = threadIdx.x >> 5;
  int lane = threadIdx.x & 31;
  int half = lane >> 4, col = lane & 15;
  int w = blockIdx.x * 8 + wid;
  int bh = w >> 7;            // 128 q-tiles per (b,h)
  int qt = w & 127;
  int q0 = qt << 4;
  int b = bh >> 4, h = bh & 15;
  bf16* lp = ldsP + wid * (16 * 32);

  const bf16* qbase = qr + ((size_t)bh * Ssz + q0) * 64;
  v16bf qa0 = load_a_frag(qbase, 64);        // d 0..31
  v16bf qa1 = load_a_frag(qbase + 32, 64);   // d 32..63

  float4 pq[8];
#pragma unroll
  for (int r = 0; r < 8; ++r) {
    int row = half * 8 + r;
    pq[r] = *(const float4*)(Pp + ((size_t)bh * Ssz + q0 + row) * 4);
  }

  float mrow[8], lrow[8];
  v8f o0 = {}, o1 = {}, o2 = {}, o3 = {};
#pragma unroll
  for (int r = 0; r < 8; ++r) { mrow[r] = -1e30f; lrow[r] = 0.f; }

  for (int j0 = 0; j0 <= q0 + 15; j0 += 32) {
    const bf16* kb = kr + ((size_t)bh * Ssz + j0) * 64;
    v16bf kb00 = load_b_frag(kb, 64);                 // keys j0..+15, d 0..31
    v16bf kb01 = load_b_frag(kb + 32, 64);            // keys j0..+15, d 32..63
    v16bf kb10 = load_b_frag(kb + 16 * 64, 64);       // keys j0+16.., d 0..31
    v16bf kb11 = load_b_frag(kb + 16 * 64 + 32, 64);

    v8f s0 = {}, s1 = {};
    s0 = wmma_bf16(qa0, kb00, s0);
    s0 = wmma_bf16(qa1, kb01, s0);
    s1 = wmma_bf16(qa0, kb10, s1);
    s1 = wmma_bf16(qa1, kb11, s1);

    float4 pk0 = *(const float4*)(Pp + ((size_t)bh * Ssz + j0 + col) * 4);
    float4 pk1 = *(const float4*)(Pp + ((size_t)bh * Ssz + j0 + 16 + col) * 4);

#pragma unroll
    for (int r = 0; r < 8; ++r) {
      int qg = q0 + half * 8 + r;
      float lw0 = __logf(pq[r].x * pk0.x + pq[r].y * pk0.y +
                         pq[r].z * pk0.z + pq[r].w * pk0.w + 1e-8f);
      float lw1 = __logf(pq[r].x * pk1.x + pq[r].y * pk1.y +
                         pq[r].z * pk1.z + pq[r].w * pk1.w + 1e-8f);
      float v0 = s0[r] * SCL + lw0;
      float v1 = s1[r] * SCL + lw1;
      v0 = (j0 + col > qg) ? -1e30f : v0;
      v1 = (j0 + 16 + col > qg) ? -1e30f : v1;

      float mloc = rmax16(fmaxf(v0, v1));
      float mnew = fmaxf(mrow[r], mloc);
      float e0 = __expf(v0 - mnew);
      float e1 = __expf(v1 - mnew);
      float sf = __expf(mrow[r] - mnew);
      lrow[r] = lrow[r] * sf + rsum16(e0 + e1);
      mrow[r] = mnew;
      o0[r] = o0[r] * sf; o1[r] = o1[r] * sf;
      o2[r] = o2[r] * sf; o3[r] = o3[r] * sf;

      int li = (half * 8 + r) * 32 + col;   // C-layout -> LDS row-major 16x32
      lp[li] = (bf16)e0;
      lp[li + 16] = (bf16)e1;
    }
    asm volatile("s_wait_dscnt 0" ::: "memory");
    v16bf pa = load_a_frag(lp, 32);          // exp(P) as A 16x32

#pragma unroll
    for (int t = 0; t < 4; ++t) {
      const bf16* vb = vt + ((size_t)bh * 64 + t * 16) * Ssz + j0;  // Vt[d][s]
      v16bf bfr = load_b_frag(vb, Ssz);
      v8f acc = (t == 0) ? o0 : (t == 1) ? o1 : (t == 2) ? o2 : o3;
      acc = wmma_bf16(pa, bfr, acc);
      if (t == 0) o0 = acc; else if (t == 1) o1 = acc;
      else if (t == 2) o2 = acc; else o3 = acc;
    }
  }

#pragma unroll
  for (int r = 0; r < 8; ++r) {
    int qg = q0 + half * 8 + r;
    float invl = 1.f / lrow[r];
    float* op = aout + ((size_t)b * Ssz + qg) * Esz + h * 64;
    op[col]      = o0[r] * invl;
    op[16 + col] = o1[r] * invl;
    op[32 + col] = o2[r] * invl;
    op[48 + col] = o3[r] * invl;
  }
}

// ---------------- host launch ----------------

extern "C" void kernel_launch(void* const* d_in, const int* in_sizes, int n_in,
                              void* d_out, int out_size, void* d_ws, size_t ws_size,
                              hipStream_t stream) {
  (void)in_sizes; (void)n_in; (void)out_size; (void)ws_size;
  const float* x  = (const float*)d_in[0];
  const float* Wq = (const float*)d_in[1];
  const float* Wk = (const float*)d_in[2];
  const float* Wv = (const float*)d_in[3];
  const float* Wr = (const float*)d_in[4];
  const float* Wo = (const float*)d_in[5];
  float* out = (float*)d_out;

  const size_t NX = (size_t)Bsz * Ssz * Esz;  // 4M elements
  char* ws = (char*)d_ws;
  size_t off = 0;
  auto take = [&](size_t bytes) -> char* {
    char* p = ws + off;
    off += (bytes + 255) & ~(size_t)255;
    return p;
  };
  bf16*  xbf  = (bf16*)take(NX * 2);
  bf16*  WqT  = (bf16*)take((size_t)Esz * Esz * 2);
  bf16*  WkT  = (bf16*)take((size_t)Esz * Esz * 2);
  bf16*  WvT  = (bf16*)take((size_t)Esz * Esz * 2);
  bf16*  WrT  = (bf16*)take((size_t)Esz * 64 * 2);
  bf16*  WoT  = (bf16*)take((size_t)Esz * Esz * 2);
  float* Qf   = (float*)take(NX * 4);
  float* Kf   = (float*)take(NX * 4);
  float* Vf   = (float*)take(NX * 4);
  float* Lg   = (float*)take((size_t)Bsz * Hn * Ssz * Kn * 4);
  float* Pb   = (float*)take((size_t)Bsz * Hn * Ssz * Kn * 4);
  bf16*  qrb  = (bf16*)take(NX * 2);
  bf16*  krb  = (bf16*)take(NX * 2);
  bf16*  vtb  = (bf16*)take(NX * 2);
  float* aout = (float*)take(NX * 4);
  bf16*  abf  = (bf16*)take(NX * 2);

  const int M = Bsz * Ssz;  // 4096

  // 1) conversions
  k_f32_to_bf16<<<(int)(NX / 256), 256, 0, stream>>>(x, xbf, (int)NX);
  k_transpose_cvt<<<(Esz * Esz) / 256, 256, 0, stream>>>(Wq, WqT, Esz, Esz);
  k_transpose_cvt<<<(Esz * Esz) / 256, 256, 0, stream>>>(Wk, WkT, Esz, Esz);
  k_transpose_cvt<<<(Esz * Esz) / 256, 256, 0, stream>>>(Wv, WvT, Esz, Esz);
  k_transpose_cvt<<<(Esz * 64) / 256, 256, 0, stream>>>(Wr, WrT, Esz, 64);
  k_transpose_cvt<<<(Esz * Esz) / 256, 256, 0, stream>>>(Wo, WoT, Esz, Esz);

  // 2) projections (WMMA): Q/K/V -> [B,H,S,64], router logits -> [B,H,S,4]
  int blkQ = (M / 32) * (Esz / 32) / 8;  // 512
  k_gemm_bf16<<<blkQ, 256, 0, stream>>>(xbf, WqT, Qf, M, Esz, Esz, 0, Dh);
  k_gemm_bf16<<<blkQ, 256, 0, stream>>>(xbf, WkT, Kf, M, Esz, Esz, 0, Dh);
  k_gemm_bf16<<<blkQ, 256, 0, stream>>>(xbf, WvT, Vf, M, Esz, Esz, 0, Dh);
  int blkR = (M / 32) * (64 / 32) / 8;   // 32
  k_gemm_bf16<<<blkR, 256, 0, stream>>>(xbf, WrT, Lg, M, 64, Esz, 0, Kn);

  // 3) routing softmax + cumsum + fractional RoPE + V transpose
  k_route_rope<<<Bsz * Hn, 256, 0, stream>>>(Qf, Kf, Vf, Lg, qrb, krb, vtb, Pb);

  // 4) causal flash attention (WMMA)
  int blkA = (Bsz * Hn * (Ssz / 16)) / 8;  // 512
  k_attn<<<blkA, 256, 0, stream>>>(qrb, krb, vtb, Pb, aout);

  // 5) output projection (WMMA)
  k_f32_to_bf16<<<(int)(NX / 256), 256, 0, stream>>>(aout, abf, (int)NX);
  k_gemm_bf16<<<blkQ, 256, 0, stream>>>(abf, WoT, out, M, Esz, Esz, 1, Dh);
}
